// SequentialLayers_44014824849870
// MI455X (gfx1250) — compile-verified, matrix operand order checked by reading, not run on hardware
//
#include <hip/hip_runtime.h>
#include <hip/hip_bf16.h>
#include <stdint.h>

// ---------------------------------------------------------------------------
// Problem constants (from reference): B=16, S=2048, D=1024, FF=4096, DT=512
// M = B*S = 32768.
// ---------------------------------------------------------------------------
#define GM   32768
#define GD   1024
#define GFF  4096
#define GS   2048
#define GDT  512

#define BM   128
#define BN   128
#define BK   64
#define LDA_S 72   // padded LDS stride (bf16 elements); multiple of 8 keeps 16B align
#define LDB_S 72

typedef __attribute__((ext_vector_type(16))) __bf16 v16bf;
typedef __attribute__((ext_vector_type(8)))  float  v8f;

union FragBF { v16bf v; uint4 q[2]; };
union AccF   { v8f   v; float f[8]; };

__device__ __forceinline__ unsigned short f2bf(float f) {
    union { float f; unsigned u; } v; v.f = f;
    unsigned r = v.u + 0x7FFFu + ((v.u >> 16) & 1u);   // round-to-nearest-even
    return (unsigned short)(r >> 16);
}

// Branchless tanh-gelu: gelu(x) = x * sigmoid(2 * 0.79788456*(x + 0.044715 x^3)).
// Lowers to v_exp_f32 (TRANS) + rcp; saturates correctly at +/-inf.
__device__ __forceinline__ float gelu_tanh(float x) {
    float u = 0.7978845608028654f * (x + 0.044715f * x * x * x);
    float e = __expf(-2.0f * u);
    return x * __builtin_amdgcn_rcpf(1.0f + e);
}

// ---------------------------------------------------------------------------
// Tiled bf16 WMMA GEMM:  D = op(A @ Bw)
//   FIRST=true :  A fp32 [M,K], epilogue gelu, D bf16 [M,N]   (X @ W1)
//   FIRST=false:  A bf16 [M,K], no act,       D fp32 [M,N]   (H @ W2)
// Bw is always fp32 row-major [K,N], converted to bf16 on the staging path.
// Block = 128 threads = 4 waves; wave tile 64x64 (4x4 WMMA tiles) ->
// 16 ds_load_b128 per 16 v_wmma per K-step (1:1), 4x fragment reuse.
// ---------------------------------------------------------------------------
template <bool FIRST>
__global__ __launch_bounds__(128)
void gemm_wmma(const void* __restrict__ Aptr, const float* __restrict__ Bw,
               void* __restrict__ Dptr, int M, int N, int K)
{
    __shared__ __align__(16) unsigned short lds_a[BM * LDA_S];
    __shared__ __align__(16) unsigned short lds_b[BN * LDB_S];

    const int tid  = threadIdx.x;
    const int lane = tid & 31;
    const int wv   = tid >> 5;      // wave 0..3
    const int wm   = wv >> 1;       // 0..1 : 64-row strip
    const int wn   = wv & 1;        // 0..1 : 64-col strip
    const int half = lane >> 4;     // lane group (K-chunk select)
    const int r    = lane & 15;     // row (A) / col (B,C)

    const int m0 = blockIdx.y * BM;
    const int n0 = blockIdx.x * BN;

    AccF acc[4][4];
    #pragma unroll
    for (int i = 0; i < 4; ++i)
        #pragma unroll
        for (int j = 0; j < 4; ++j)
            #pragma unroll
            for (int e = 0; e < 8; ++e)
                acc[i][j].f[e] = 0.0f;

    for (int kt = 0; kt < K; kt += BK) {
        // ---- stage A tile (BM x BK) into LDS as bf16 ----
        if (FIRST) {
            const float* A = (const float*)Aptr;
            #pragma unroll
            for (int i = 0; i < 16; ++i) {
                int idx = tid + i * 128;                 // 0..2047 float4's
                int row = idx >> 4;                      // BK/4 = 16 per row
                int c4  = idx & 15;
                float4 f = *(const float4*)(A + (size_t)(m0 + row) * K + kt + c4 * 4);
                uint2 p;
                p.x = (unsigned)f2bf(f.x) | ((unsigned)f2bf(f.y) << 16);
                p.y = (unsigned)f2bf(f.z) | ((unsigned)f2bf(f.w) << 16);
                *(uint2*)&lds_a[row * LDA_S + c4 * 4] = p;
            }
        } else {
            const unsigned short* A = (const unsigned short*)Aptr;
            #pragma unroll
            for (int i = 0; i < 8; ++i) {
                int idx = tid + i * 128;                 // 0..1023 uint4's
                int row = idx >> 3;                      // BK/8 = 8 per row
                int c8  = idx & 7;
                uint4 q = *(const uint4*)(A + (size_t)(m0 + row) * K + kt + c8 * 8);
                *(uint4*)&lds_a[row * LDA_S + c8 * 8] = q;
            }
        }
        // ---- stage B tile (BK x BN) into LDS column-major (n-major) bf16 ----
        #pragma unroll
        for (int i = 0; i < 16; ++i) {
            int idx = tid + i * 128;                     // 0..2047 float4's
            int k   = idx >> 5;                          // BN/4 = 32 per k-row
            int c4  = idx & 31;
            float4 f = *(const float4*)(Bw + (size_t)(kt + k) * N + n0 + c4 * 4);
            lds_b[(c4 * 4 + 0) * LDB_S + k] = f2bf(f.x);
            lds_b[(c4 * 4 + 1) * LDB_S + k] = f2bf(f.y);
            lds_b[(c4 * 4 + 2) * LDB_S + k] = f2bf(f.z);
            lds_b[(c4 * 4 + 3) * LDB_S + k] = f2bf(f.w);
        }
        __syncthreads();

        // ---- 2 WMMA K-steps of 32 ----
        #pragma unroll
        for (int ks = 0; ks < 2; ++ks) {
            const int k0 = ks * 32 + half * 8;           // ISA 16-bit A layout:
            FragBF a[4], b[4];                           // chunks K=[k0..k0+7],[k0+16..k0+23]
            #pragma unroll
            for (int mi = 0; mi < 4; ++mi) {
                int row = wm * 64 + mi * 16 + r;
                a[mi].q[0] = *(const uint4*)&lds_a[row * LDA_S + k0];
                a[mi].q[1] = *(const uint4*)&lds_a[row * LDA_S + k0 + 16];
            }
            #pragma unroll
            for (int ni = 0; ni < 4; ++ni) {
                int col = wn * 64 + ni * 16 + r;
                b[ni].q[0] = *(const uint4*)&lds_b[col * LDB_S + k0];
                b[ni].q[1] = *(const uint4*)&lds_b[col * LDB_S + k0 + 16];
            }
            #pragma unroll
            for (int mi = 0; mi < 4; ++mi)
                #pragma unroll
                for (int ni = 0; ni < 4; ++ni)
                    acc[mi][ni].v = __builtin_amdgcn_wmma_f32_16x16x32_bf16(
                        false, a[mi].v, false, b[ni].v,
                        (short)0, acc[mi][ni].v, false, false);
        }
        __syncthreads();
    }

    // ---- epilogue: C/D layout = lane(r,half), VGPR v -> (row v+8*half, col r) ----
    #pragma unroll
    for (int mi = 0; mi < 4; ++mi) {
        #pragma unroll
        for (int ni = 0; ni < 4; ++ni) {
            #pragma unroll
            for (int v = 0; v < 8; ++v) {
                size_t row = (size_t)(m0 + wm * 64 + mi * 16 + v + 8 * half);
                size_t col = (size_t)(n0 + wn * 64 + ni * 16 + r);
                float x = acc[mi][ni].f[v];
                if (FIRST) {
                    ((unsigned short*)Dptr)[row * (size_t)N + col] = f2bf(gelu_tanh(x));
                } else {
                    ((float*)Dptr)[row * (size_t)N + col] = x;
                }
            }
        }
    }
}

// ---------------------------------------------------------------------------
// Tiny intervention: 16 rows x 512 through two 512x512 matmuls (~17 MFLOP).
// ---------------------------------------------------------------------------
__global__ void intervene_stage1(const float* __restrict__ out, const int* __restrict__ eot,
                                 const float* __restrict__ Wrot, float* __restrict__ tmp)
{
    int gid = blockIdx.x * blockDim.x + threadIdx.x;   // 16*512 threads
    int b = gid >> 9;
    int n = gid & 511;
    const float* row = out + ((size_t)b * GS + (size_t)eot[b]) * GD;  // ST=0
    float s = 0.0f;
    for (int k = 0; k < GDT; ++k)
        s = fmaf(row[k], Wrot[(size_t)k * GDT + n], s);
    tmp[(size_t)b * GDT + n] = s;
}

__global__ void intervene_stage2(float* __restrict__ out, const int* __restrict__ eot,
                                 const float* __restrict__ Winv, const float* __restrict__ tmp)
{
    int gid = blockIdx.x * blockDim.x + threadIdx.x;
    int b = gid >> 9;
    int n = gid & 511;
    float s = 0.0f;
    for (int k = 0; k < GDT; ++k)
        s = fmaf(tmp[(size_t)b * GDT + k], Winv[(size_t)k * GDT + n], s);
    out[((size_t)b * GS + (size_t)eot[b]) * GD + n] = s;
}

// ---------------------------------------------------------------------------
extern "C" void kernel_launch(void* const* d_in, const int* in_sizes, int n_in,
                              void* d_out, int out_size, void* d_ws, size_t ws_size,
                              hipStream_t stream)
{
    const float* X   = (const float*)d_in[0];   // hidden_states [16,2048,1024] f32
    const float* W1  = (const float*)d_in[1];   // [1024,4096] f32
    const float* W2  = (const float*)d_in[2];   // [4096,1024] f32
    const float* Wr  = (const float*)d_in[3];   // [512,512] f32
    const float* Wi  = (const float*)d_in[4];   // [512,512] f32
    const int*   eot = (const int*)d_in[5];     // [16] i32
    float* out = (float*)d_out;                 // [16,2048,1024] f32

    // workspace: H (bf16, M x FF) then tmp (f32, 16 x 512)
    unsigned short* H = (unsigned short*)d_ws;
    float* tmp = (float*)((char*)d_ws + (size_t)GM * GFF * sizeof(unsigned short));

    dim3 blk(128, 1, 1);

    // GEMM1: H = gelu(X @ W1), bf16 out
    gemm_wmma<true><<<dim3(GFF / BN, GM / BM), blk, 0, stream>>>(
        (const void*)X, W1, (void*)H, GM, GFF, GD);

    // GEMM2: out = H @ W2, f32 out
    gemm_wmma<false><<<dim3(GD / BN, GM / BM), blk, 0, stream>>>(
        (const void*)H, W2, (void*)out, GM, GD, GFF);

    // intervention on the 16 eot rows, cols [0,512)
    intervene_stage1<<<(16 * GDT) / 256, 256, 0, stream>>>(out, eot, Wr, tmp);
    intervene_stage2<<<(16 * GDT) / 256, 256, 0, stream>>>(out, eot, Wi, tmp);
}
